// LocalAggregator_68453188764260
// MI455X (gfx1250) — matrix operand order repaired and checked by reading, not compiled, not run on hardware
//
#include <hip/hip_runtime.h>

typedef __bf16 v16bf __attribute__((ext_vector_type(16)));
typedef float  v8f   __attribute__((ext_vector_type(8)));
typedef unsigned int v8u __attribute__((ext_vector_type(8)));

namespace {
constexpr int kB = 512;
constexpr int kN = 200;
constexpr int kD = 128;
constexpr int kNP = 224;      // padded N (multiple of 32 for K-chunking)
constexpr int kHS = 132;      // LDS row stride (bf16 elems): 264B, de-skews the 64 banks
constexpr int kTilesI = 13;   // ceil(200/16)

// Dynamic LDS layout (bytes)
constexpr int OFF_HB = 0;                       // ushort[kNP][kHS]      h[b] as bf16 (padded rows = 0)
constexpr int OFF_HS = OFF_HB + kNP * kHS * 2;  // ushort[4*16][kHS]     (h_i ⊙ a_k) row tiles, bf16
constexpr int OFF_SC = OFF_HS + 64 * kHS * 2;   // float [16][kNP]       score panel
constexpr int OFF_AL = OFF_SC + 16 * kNP * 4;   // ushort[16][kNP]       softmaxed alpha, bf16
constexpr int LDS_BYTES = OFF_AL + 16 * kNP * 2;
}

__device__ __forceinline__ unsigned short f2bf(float f) {
  unsigned int u = __float_as_uint(f);
  u += 0x7FFFu + ((u >> 16) & 1u);   // round-to-nearest-even
  return (unsigned short)(u >> 16);
}

__global__ void __launch_bounds__(256)
local_aggregator_kernel(const float* __restrict__ h,
                        const int* __restrict__ adj,
                        const float* __restrict__ a0,
                        const float* __restrict__ a1,
                        const float* __restrict__ a2,
                        const float* __restrict__ a3,
                        float* __restrict__ out) {
  extern __shared__ char smem[];
  unsigned short* hB = (unsigned short*)(smem + OFF_HB);
  unsigned short* hS = (unsigned short*)(smem + OFF_HS);
  float*          Sc = (float*)(smem + OFF_SC);
  unsigned short* Al = (unsigned short*)(smem + OFF_AL);

  const int b  = blockIdx.x / kTilesI;
  const int it = blockIdx.x % kTilesI;
  const int i0 = it * 16;

  const int tid  = threadIdx.x;
  const int lane = tid & 31;
  const int wave = tid >> 5;
  const int nIdx = lane & 15;   // N-index (B/C/D) or M-index (A)
  const int half = lane >> 4;   // lane group 0-15 / 16-31

  const float negHuge = -9.0e15f;
  const float negInf  = __uint_as_float(0xff800000u);

  // ---------- Phase 0: stage h[b] (bf16), scaled A rows, init pad columns ----------
  for (int idx = tid; idx < kNP * kD; idx += 256) {
    const int r = idx >> 7, d = idx & 127;
    const float v = (r < kN) ? h[((size_t)b * kN + r) * kD + d] : 0.f;
    hB[r * kHS + d] = f2bf(v);
  }
  for (int idx = tid; idx < 4 * 16 * kD; idx += 256) {
    const int k = idx >> 11;
    const int i = (idx >> 7) & 15;
    const int d = idx & 127;
    const float av = (k == 0) ? a0[d] : (k == 1) ? a1[d] : (k == 2) ? a2[d] : a3[d];
    const int gi = i0 + i;
    const float v = (gi < kN) ? h[((size_t)b * kN + gi) * kD + d] * av : 0.f;
    hS[(k * 16 + i) * kHS + d] = f2bf(v);
  }
  // Only cols 208..223 need init: phase 1 writes cols 0..207 unconditionally.
  {
    const int row = tid >> 4;
    const int col = 208 + (tid & 15);
    Sc[row * kNP + col] = negInf;
  }
  __syncthreads();

  // ---------- Phase 1: relation score tiles via WMMA, branchless masked select ----------
  for (int jt = wave; jt < kTilesI; jt += 8) {
    const int j0 = jt * 16;
    const int j  = j0 + nIdx;
    // B fragments: B[K=d][N=j] = h[j0+j][d], four K-chunks of 32
    v16bf Bf[4];
    {
      const unsigned short* hrow = hB + j * kHS;
      #pragma unroll
      for (int kc = 0; kc < 4; ++kc) {
        v8u w;
        #pragma unroll
        for (int v = 0; v < 8; ++v) {
          const int kk = kc * 32 + 16 * (v >> 2) + 8 * half + 2 * (v & 3);      // K-pair per VGPR
          w[v] = *(const unsigned int*)(hrow + kk);
        }
        Bf[kc] = __builtin_bit_cast(v16bf, w);
      }
    }
    // Branchless adj tile load: clamp address, mask value.
    int adjv[8];
    {
      const int jj = (j < kN) ? j : (kN - 1);
      #pragma unroll
      for (int r = 0; r < 8; ++r) {
        const int i  = i0 + r + 8 * half;
        const int ii = (i < kN) ? i : (kN - 1);
        const int av = adj[((size_t)b * kN + ii) * kN + jj];
        adjv[r] = (i < kN && j < kN) ? av : 0;
      }
    }
    // Selected-score panel: default no-edge value; pad cols keep -inf.
    float sel[8];
    const float selInit = (j < kN) ? negHuge : negInf;
    #pragma unroll
    for (int r = 0; r < 8; ++r) sel[r] = selInit;

    #pragma unroll
    for (int k = 0; k < 4; ++k) {
      v8f acc = {};
      const unsigned short* arow = hS + (k * 16 + nIdx) * kHS;
      #pragma unroll
      for (int kc = 0; kc < 4; ++kc) {
        v8u w;
        #pragma unroll
        for (int v = 0; v < 8; ++v) {
          const int kk = kc * 32 + 16 * (v >> 2) + 8 * half + 2 * (v & 3);
          w[v] = *(const unsigned int*)(arow + kk);
        }
        v16bf Af = __builtin_bit_cast(v16bf, w);
        acc = __builtin_amdgcn_wmma_f32_16x16x32_bf16(false, Af, false, Bf[kc],
                                                      (short)0, acc, false, false);
      }
      #pragma unroll
      for (int r = 0; r < 8; ++r) {
        float s = acc[r];
        s = (s > 0.f) ? s : 0.2f * s;                      // leaky relu (v_cndmask)
        sel[r] = (adjv[r] == k + 1) ? s : sel[r];          // predicated select, no branch
      }
    }
    #pragma unroll
    for (int r = 0; r < 8; ++r)
      Sc[(r + 8 * half) * kNP + j] = sel[r];               // unconditional stores
  }
  __syncthreads();

  // ---------- Phase 2: row softmax (16 threads per row, width-16 shuffles) ----------
  {
    const int row = tid >> 4;
    const int c   = tid & 15;
    float* Srow = Sc + row * kNP;
    float m = negInf;
    for (int col = c; col < kNP; col += 16) m = fmaxf(m, Srow[col]);
    #pragma unroll
    for (int off = 8; off > 0; off >>= 1) m = fmaxf(m, __shfl_xor(m, off, 16));
    float sum = 0.f;
    for (int col = c; col < kNP; col += 16) {
      const float e = __expf(Srow[col] - m);
      Srow[col] = e;
      sum += e;
    }
    #pragma unroll
    for (int off = 8; off > 0; off >>= 1) sum += __shfl_xor(sum, off, 16);
    const float inv = 1.0f / sum;
    for (int col = c; col < kNP; col += 16)
      Al[row * kNP + col] = f2bf(Srow[col] * inv);
  }
  __syncthreads();

  // ---------- Phase 3: out tile = alpha[16,224] x h[224,128]; one d-tile per wave ----------
  {
    const int d0 = wave * 16;
    v8f acc = {};
    const unsigned short* prow = Al + nIdx * kNP;
    #pragma unroll
    for (int kc = 0; kc < 7; ++kc) {
      v8u wa;
      #pragma unroll
      for (int v = 0; v < 8; ++v) {
        const int kk = kc * 32 + 16 * (v >> 2) + 8 * half + 2 * (v & 3);
        wa[v] = *(const unsigned int*)(prow + kk);
      }
      v16bf Af = __builtin_bit_cast(v16bf, wa);
      v8u wb;
      #pragma unroll
      for (int v = 0; v < 8; ++v) {
        const int kr = kc * 32 + 16 * (v >> 2) + 8 * half + 2 * (v & 3);        // K rows of h, column d0+nIdx
        const unsigned int lo = hB[kr * kHS + d0 + nIdx];
        const unsigned int hi = hB[(kr + 1) * kHS + d0 + nIdx];
        wb[v] = lo | (hi << 16);
      }
      v16bf Bf = __builtin_bit_cast(v16bf, wb);
      acc = __builtin_amdgcn_wmma_f32_16x16x32_bf16(false, Af, false, Bf,
                                                    (short)0, acc, false, false);
    }
    #pragma unroll
    for (int r = 0; r < 8; ++r) {
      const int i = i0 + r + 8 * half;
      if (i < kN) out[((size_t)b * kN + i) * kD + d0 + nIdx] = acc[r];
    }
  }
}

extern "C" void kernel_launch(void* const* d_in, const int* in_sizes, int n_in,
                              void* d_out, int out_size, void* d_ws, size_t ws_size,
                              hipStream_t stream) {
  const float* h   = (const float*)d_in[0];
  const int*   adj = (const int*)d_in[1];
  const float* a0  = (const float*)d_in[2];
  const float* a1  = (const float*)d_in[3];
  const float* a2  = (const float*)d_in[4];
  const float* a3  = (const float*)d_in[5];
  float* out = (float*)d_out;

  dim3 grid(kB * kTilesI);   // one block per (batch, 16-row tile)
  dim3 block(256);           // 8 wave32 waves
  local_aggregator_kernel<<<grid, block, LDS_BYTES, stream>>>(h, adj, a0, a1, a2, a3, out);
}
